// RNNModel_9981503996347
// MI455X (gfx1250) — matrix operand order repaired and checked by reading, not compiled, not run on hardware
//
#include <hip/hip_runtime.h>
#include <hip/hip_bf16.h>

#define INPUT  64
#define HIDDEN 128
#define BATCH  512
#define SEQ    512

#define HPAD 136   // bf16 elems per h row: 128 + 8 pad -> 272B row = conflict-free b128 gathers
#define TPAD 72    // bf16 elems per token row: 64 + 8 pad -> 144B row

typedef __attribute__((ext_vector_type(16))) __bf16       v16bf;
typedef __attribute__((ext_vector_type(8)))  float        v8f;
typedef __attribute__((ext_vector_type(4)))  unsigned int u32x4;
typedef __attribute__((ext_vector_type(4)))  float        f32x4;

union BfTile {
    v16bf v;
    u32x4 q[2];
    unsigned short s[16];
};

__device__ __forceinline__ unsigned short f2bf(float f) {
    unsigned int u = __float_as_uint(f);
    unsigned int r = (u + 0x7FFFu + ((u >> 16) & 1u)) >> 16;   // round-to-nearest-even
    return (unsigned short)r;
}
__device__ __forceinline__ float bf2f(unsigned short h) {
    return __uint_as_float(((unsigned int)h) << 16);
}

// Branch-free tanh: 1 - 2/(exp2(2*log2e*x)+1).
// Lowers to v_mul + v_exp_f32 + v_add + v_rcp_f32 + v_fma; saturates to +/-1,
// no EXEC-divergent slow path on the 512-step critical chain.
__device__ __forceinline__ float fast_tanh(float x) {
    float e = __builtin_amdgcn_exp2f(x * 2.8853900817779268f);  // exp(2x)
    return 1.0f - 2.0f * __builtin_amdgcn_rcpf(e + 1.0f);
}

__global__ __launch_bounds__(256) void rnn_fused_kernel(
    const float* __restrict__ token, const float* __restrict__ W_ih,
    const float* __restrict__ W_hh,  const float* __restrict__ b_ih,
    const float* __restrict__ b_hh,  const float* __restrict__ W_lin,
    const float* __restrict__ b_lin, float* __restrict__ out)
{
    __shared__ unsigned short h_lds[2][16 * HPAD];     // bf16 hidden state, double buffered
    __shared__ unsigned short tok_lds[2][16 * TPAD];   // bf16 token tile, double buffered

    const int tid   = threadIdx.x;
    const int lane  = tid & 31;
    const int wave  = tid >> 5;            // 0..7 : owns N-tile [16*wave, 16*wave+16)
    const int b0    = blockIdx.x * 16;     // batch tile base

    // A-operand (this code's contiguous k order, 16x32): lanes 0-15 row M=lane, K 0..15;
    // lanes 16-31 row M=lane-16, K 16..31 (matches B)
    const int mrow  = lane & 15;
    const int khalf = (lane >> 4) ? 8 : 0;
    // B-operand (this code's contiguous k order, same as A): lane column n = lane&15; lanes 0-15 hold K 0..15,
    // lanes 16-31 hold K 16..31 (2 per VGPR, consecutive)
    const int kb16  = (lane >> 4) ? 16 : 0;
    const int ncol  = wave * 16 + (lane & 15);   // global hidden column for B / C / D

    // ---- One-time: load weight B-tiles into registers as bf16 ----
    BfTile Bh[4];   // W_hh^T, K = 128 -> 4 K-tiles of 32.  B(k,n) = W_hh[n][k]
    #pragma unroll
    for (int kt = 0; kt < 4; ++kt) {
        const float* src = W_hh + ncol * HIDDEN + kt * 32 + kb16;
        #pragma unroll
        for (int j = 0; j < 16; ++j) Bh[kt].s[j] = f2bf(src[j]);
    }
    BfTile Bx[2];   // W_ih^T, K = 64 -> 2 K-tiles of 32.  B(k,n) = W_ih[n][k]
    #pragma unroll
    for (int kt = 0; kt < 2; ++kt) {
        const float* src = W_ih + ncol * INPUT + kt * 32 + kb16;
        #pragma unroll
        for (int j = 0; j < 16; ++j) Bx[kt].s[j] = f2bf(src[j]);
    }
    const float bias = b_ih[ncol] + b_hh[ncol];

    // ---- h0 = 0, preload token tile for t = 0 ----
    for (int i = tid; i < 16 * HPAD; i += 256) { h_lds[0][i] = 0; h_lds[1][i] = 0; }

    const long grow   = (long)(b0 + (tid >> 4)) * (SEQ * INPUT);  // this thread's token row
    const int  pf_col = (tid & 15) * 4;
    {
        const f32x4 v = *(const f32x4*)(token + grow + 0 * INPUT + pf_col);
        unsigned short* dst = &tok_lds[0][(tid >> 4) * TPAD + pf_col];
        dst[0] = f2bf(v.x); dst[1] = f2bf(v.y); dst[2] = f2bf(v.z); dst[3] = f2bf(v.w);
    }
    __syncthreads();

    const int mbase = (lane >> 4) ? 8 : 0;   // D layout: VGPR r -> row r (+8 for upper lanes)

    // ---- Sequential scan: h = tanh(x_t @ W_ih^T + bias + h @ W_hh^T) ----
    for (int t = 0; t < SEQ; ++t) {
        const int p = t & 1;

        // Prefetch next step's token tile (issue global load early, consume late)
        const int tn = (t + 1 < SEQ) ? (t + 1) : t;
        const f32x4 pf = *(const f32x4*)(token + grow + tn * INPUT + pf_col);

        // Token path: acc1 = bias + x @ W_ih^T  (2 chained WMMAs)
        v8f acc1;
        #pragma unroll
        for (int r = 0; r < 8; ++r) acc1[r] = bias;
        #pragma unroll
        for (int kt = 0; kt < 2; ++kt) {
            BfTile a;
            const unsigned short* base = &tok_lds[p][mrow * TPAD + kt * 32 + kb16];
            a.q[0] = *(const u32x4*)(base);
            a.q[1] = *(const u32x4*)(base + 8);
            acc1 = __builtin_amdgcn_wmma_f32_16x16x32_bf16(
                false, a.v, false, Bx[kt].v, (short)0, acc1, false, false);
        }

        // Hidden path: acc2 = h @ W_hh^T  (independent chain of 4 WMMAs)
        v8f acc2 = {};
        #pragma unroll
        for (int kt = 0; kt < 4; ++kt) {
            BfTile a;
            const unsigned short* base = &h_lds[p][mrow * HPAD + kt * 32 + kb16];
            a.q[0] = *(const u32x4*)(base);
            a.q[1] = *(const u32x4*)(base + 8);
            acc2 = __builtin_amdgcn_wmma_f32_16x16x32_bf16(
                false, a.v, false, Bh[kt].v, (short)0, acc2, false, false);
        }

        // h_new = tanh(acc1 + acc2)  -> bf16 into the other h buffer
        {
            unsigned short* hw = &h_lds[1 - p][0];
            #pragma unroll
            for (int r = 0; r < 8; ++r) {
                float v = fast_tanh(acc1[r] + acc2[r]);
                hw[(mbase + r) * HPAD + ncol] = f2bf(v);
            }
        }

        // Commit prefetched token tile to the other token buffer
        {
            unsigned short* dst = &tok_lds[1 - p][(tid >> 4) * TPAD + pf_col];
            dst[0] = f2bf(pf.x); dst[1] = f2bf(pf.y); dst[2] = f2bf(pf.z); dst[3] = f2bf(pf.w);
        }
        __syncthreads();
    }

    // ---- Head: out[b] = h_last @ W_lin^T + b_lin   (h_last lives in h_lds[SEQ & 1]) ----
    if (tid < 16) {
        const unsigned short* hr = &h_lds[SEQ & 1][tid * HPAD];
        float s = b_lin[0];
        #pragma unroll 8
        for (int n = 0; n < HIDDEN; ++n) s += bf2f(hr[n]) * W_lin[n];
        out[b0 + tid] = s;
    }
}

extern "C" void kernel_launch(void* const* d_in, const int* in_sizes, int n_in,
                              void* d_out, int out_size, void* d_ws, size_t ws_size,
                              hipStream_t stream) {
    const float* token = (const float*)d_in[0];
    const float* W_ih  = (const float*)d_in[1];
    const float* W_hh  = (const float*)d_in[2];
    const float* b_ih  = (const float*)d_in[3];
    const float* b_hh  = (const float*)d_in[4];
    const float* W_lin = (const float*)d_in[5];
    const float* b_lin = (const float*)d_in[6];
    float* out = (float*)d_out;

    rnn_fused_kernel<<<BATCH / 16, 256, 0, stream>>>(
        token, W_ih, W_hh, b_ih, b_hh, W_lin, b_lin, out);
}
